// SMARTAgentDecoder_45535243272607
// MI455X (gfx1250) — compile-verified, matrix operand order checked by reading, not run on hardware
//
#include <hip/hip_runtime.h>
#include <hip/hip_bf16.h>
#include <math.h>

// ---------------------------------------------------------------------------
// SMART agent decoder forward for gfx1250 (MI455X).
// Dense contractions -> v_wmma_f32_16x16x32_f16 (wave32), LN/ReLU fused into
// A staging (one HBM pass), B tiles staged via the Tensor Data Mover
// (tensor_load_to_lds + s_wait_tensorcnt) with OOB zero-fill, fragments fed
// from LDS with 2x b128 vector loads per fragment. Segment softmax via
// f32 global atomics with order-preserving uint encoding for the max.
// ---------------------------------------------------------------------------

typedef __attribute__((ext_vector_type(16))) _Float16 v16h;
typedef __attribute__((ext_vector_type(8)))  _Float16 v8h;
typedef __attribute__((ext_vector_type(8)))  float    v8f;
typedef __attribute__((ext_vector_type(4)))  unsigned v4u;
typedef __attribute__((ext_vector_type(4)))  int      v4i;
typedef __attribute__((ext_vector_type(8)))  int      v8i;

#define A_NUM   1500
#define T_NUM   18
#define N_NUM   27000      // A*T
#define H_NUM   128
#define NH_NUM  8
#define HD_NUM  16
#define L_NUM   3
#define F_NUM   64
#define TOK_NUM 512
#define MS_NUM  36000
#define ET_NUM  150000
#define EA_NUM  200000
#define EP_NUM  200000
#define FF_NUM  512
#define CH_NUM  50000      // fourier row chunk

#define PI_F     3.14159265358979f
#define TWO_PI_F 6.28318530717959f

// gemm flags
#define GF_LN    1   // LayerNorm rows of A (over K) while staging
#define GF_ARELU 2   // ReLU on A after LN
#define GF_CRELU 4   // ReLU on output
#define GF_ACC   8   // C += result (residual / batched accumulate)

#if __has_builtin(__builtin_amdgcn_tensor_load_to_lds)
#define HAVE_TDM 1
#endif

// ---------------------------------------------------------------------------
// WMMA GEMM:  C[M x Ncol] (+)= act( LN?(A)[M x K] @ B[K x Ncol] + bias )
// 256 threads = 8 wave32 waves; block tile 16 rows x 128 cols; one 16x16
// WMMA tile per wave.  KFIX=128 -> K-loop fully unrolled (4 static WMMAs).
// ---------------------------------------------------------------------------
template<int KFIX>
__global__ __launch_bounds__(256) void k_gemm_t(
    const float* __restrict__ A, long lda, int M,
    const float* __restrict__ B, long ldb, int Krt, int Ncol,
    const float* __restrict__ bias,
    const float* __restrict__ lng, const float* __restrict__ lnb,
    float* __restrict__ C, long ldc, int flags)
{
  const int K = KFIX ? KFIX : Krt;

  __shared__ float    Af32[16][264];   // cached A rows (LN path, K<=256)
  __shared__ _Float16 As[16][40];      // 16 x 32 f16 K-tile (80B row, 16B aligned)
  __shared__ _Float16 Bs_t[128][40];   // transposed B tile: [col][k]
  __shared__ float    Bs32[32][128];   // raw f32 B tile (TDM destination)
  __shared__ float    red[256], red2[256];
  __shared__ float    mu_s[16], rs_s[16];

  const int tid  = threadIdx.x;
  const int wave = tid >> 5;
  const int lane = tid & 31;
  const int row0 = blockIdx.x * 16;
  const int col0 = blockIdx.y * 128;
  const bool useAc = (flags & GF_LN) && (K <= 256);   // cache A rows in LDS

  if (flags & GF_LN) {            // row mean/var over K (LN dim == K here)
    const int r = tid >> 4, t = tid & 15;
    float s = 0.f, q2 = 0.f;
    if (row0 + r < M) {
      for (int kx = t; kx < K; kx += 16) {
        float a = A[(long)(row0 + r) * lda + kx];
        if (useAc) Af32[r][kx] = a;
        s += a; q2 += a * a;
      }
    }
    red[tid] = s; red2[tid] = q2;
    __syncthreads();
    if (t == 0) {
      float ss = 0.f, qq = 0.f;
      for (int j = 0; j < 16; ++j) { ss += red[(r << 4) + j]; qq += red2[(r << 4) + j]; }
      float m  = ss / (float)K;
      float va = qq / (float)K - m * m;
      mu_s[r] = m;
      rs_s[r] = rsqrtf(va + 1e-5f);
    }
    __syncthreads();
  }

  v8f acc = {};
  const int lrow  = lane & 15;
  const int kbase = (lane < 16) ? 0 : 8;

  for (int k0 = 0; k0 < K; k0 += 32) {
    // ---- stage A 16x32 (f32 -> f16) with fused LN / ReLU; zero padding ----
    for (int i = tid; i < 16 * 32; i += 256) {
      int r = i >> 5, kk = i & 31;
      float v = 0.f;
      if (row0 + r < M && k0 + kk < K) {
        v = useAc ? Af32[r][k0 + kk] : A[(long)(row0 + r) * lda + k0 + kk];
        if (flags & GF_LN)    v = (v - mu_s[r]) * rs_s[r] * lng[k0 + kk] + lnb[k0 + kk];
        if (flags & GF_ARELU) v = fmaxf(v, 0.f);
      }
      As[r][kk] = (_Float16)v;
    }
#ifdef HAVE_TDM
    // ---- stage B via Tensor Data Mover: 2D tile 128(cols) x 32(rows),
    //      tensor dims = remaining extents so TDM zero-fills the padding.
    //      Scalar-branch wave0 only (TDM ignores EXEC; must be branch-skipped).
    if (tid < 32) {
      unsigned lds_off = (unsigned)(unsigned long long)(void*)&Bs32[0][0];
      unsigned long long ga = (unsigned long long)(const void*)B +
                              ((unsigned long long)k0 * (unsigned long long)ldb +
                               (unsigned long long)col0) * 4ull;
      unsigned td0 = (unsigned)(Ncol - col0);      // elements along contiguous dim
      unsigned td1 = (unsigned)(K - k0);           // rows
      v4u g0;
      g0.x = 1u;                                   // count=1 user descriptor
      g0.y = lds_off;                              // lds_addr [63:32]
      g0.z = (unsigned)ga;                         // global_addr [95:64]
      g0.w = ((unsigned)(ga >> 32) & 0x01FFFFFFu) | 0x80000000u; // addr hi | type=2
      v8i g1;
      g1[0] = (int)(2u << 16);                     // data_size=4B; no multicast/pad
      g1[1] = (int)((td0 & 0xFFFFu) << 16);        // tensor_dim0[15:0] @bit48
      g1[2] = (int)((td0 >> 16) | ((td1 & 0xFFFFu) << 16));  // td0 hi | td1 lo
      g1[3] = (int)((td1 >> 16) | (128u << 16));   // td1 hi | tile_dim0=128
      g1[4] = 32;                                  // tile_dim1=32, tile_dim2=0
      g1[5] = (int)(unsigned)ldb;                  // tensor_dim0_stride lo32
      g1[6] = 0;                                   // stride0 hi16 | stride1 lo16
      g1[7] = 0;
      v4i gz = {0, 0, 0, 0};
#if defined(__clang_major__) && (__clang_major__ >= 23)
      v8i gz8 = {0, 0, 0, 0, 0, 0, 0, 0};
      __builtin_amdgcn_tensor_load_to_lds(g0, g1, gz, gz, gz8, 0);
#else
      __builtin_amdgcn_tensor_load_to_lds(g0, g1, gz, gz, 0);
#endif
    }
    __builtin_amdgcn_s_wait_tensorcnt((short)0);
    __syncthreads();
    // cooperative f32 -> f16 transpose into fragment-friendly layout
    for (int i = tid; i < 32 * 128; i += 256) {
      int kk = i >> 7, c = i & 127;
      Bs_t[c][kk] = (_Float16)Bs32[kk][c];
    }
#else
    // ---- fallback: direct global -> LDS (transposed) staging ----
    for (int i = tid; i < 32 * 128; i += 256) {
      int kk = i >> 7, c = i & 127;
      float v = 0.f;
      if (k0 + kk < K && col0 + c < Ncol)
        v = B[(long)(k0 + kk) * ldb + col0 + c];
      if (k0 + 32 + kk < K && col0 + c < Ncol)
        __builtin_prefetch(&B[(long)(k0 + 32 + kk) * ldb + col0 + c], 0, 0);
      Bs_t[c][kk] = (_Float16)v;
    }
#endif
    __syncthreads();
    // ---- fragments: 2x ds b128 each (halves kbase..+7 and kbase+16..+23) ----
    v8h a0 = *(const v8h*)&As[lrow][kbase];
    v8h a1 = *(const v8h*)&As[lrow][kbase + 16];
    v8h b0 = *(const v8h*)&Bs_t[(wave << 4) + lrow][kbase];
    v8h b1 = *(const v8h*)&Bs_t[(wave << 4) + lrow][kbase + 16];
    v16h af = __builtin_shufflevector(a0, a1, 0,1,2,3,4,5,6,7,8,9,10,11,12,13,14,15);
    v16h bf = __builtin_shufflevector(b0, b1, 0,1,2,3,4,5,6,7,8,9,10,11,12,13,14,15);
    acc = __builtin_amdgcn_wmma_f32_16x16x32_f16(false, af, false, bf,
                                                 (short)0, acc, false, false);
    __syncthreads();
  }

  // C/D layout: VGPR r, lanes 0-15 -> M=r, N=lane ; lanes 16-31 -> M=r+8
  const int nn   = lane & 15;
  const int mofs = (lane < 16) ? 0 : 8;
#pragma unroll
  for (int r = 0; r < 8; ++r) {
    int row = row0 + mofs + r;
    int col = col0 + (wave << 4) + nn;
    if (row < M && col < Ncol) {
      float v = acc[r];
      if (bias)             v += bias[col];
      if (flags & GF_ACC)   v += C[(long)row * ldc + col];
      if (flags & GF_CRELU) v  = fmaxf(v, 0.f);
      C[(long)row * ldc + col] = v;
    }
  }
}

// ---------------------------------------------------------------------------
// small elementwise / edge kernels
// ---------------------------------------------------------------------------
__device__ __forceinline__ float wrapf(float a) {
  float r = fmodf(a + PI_F, TWO_PI_F);
  if (r < 0.f) r += TWO_PI_F;
  return r - PI_F;
}
__device__ __forceinline__ unsigned fenc(float x) {
  unsigned u = __float_as_uint(x);
  return (u & 0x80000000u) ? ~u : (u | 0x80000000u);
}
__device__ __forceinline__ float fdec(unsigned e) {
  unsigned u = (e & 0x80000000u) ? (e & 0x7fffffffu) : ~e;
  return __uint_as_float(u);
}

__global__ void k_fill(float* p, long n, float v) {
  long i = (long)blockIdx.x * blockDim.x + threadIdx.x;
  if (i < n) p[i] = v;
}
__global__ void k_copy(float* o, const float* a, long n) {
  long i = (long)blockIdx.x * blockDim.x + threadIdx.x;
  if (i < n) o[i] = a[i];
}
// fourier feature expansion f = [cos(2pi*x*fr), sin(2pi*x*fr), x], layout (rows,din,129)
__global__ void k_fgen(const float* __restrict__ X, long rows, int din,
                       const float* __restrict__ freqs, float* __restrict__ f) {
  long i = (long)blockIdx.x * blockDim.x + threadIdx.x;
  long tot = rows * din * 129;
  if (i >= tot) return;
  int j = (int)(i % 129);
  long rd = i / 129;
  int d = (int)(rd % din);
  long r = rd / din;
  float x = X[r * din + d];
  float v;
  if (j < F_NUM)            v = __cosf(x * freqs[d * F_NUM + j] * TWO_PI_F);
  else if (j < 2 * F_NUM)   v = __sinf(x * freqs[d * F_NUM + (j - F_NUM)] * TWO_PI_F);
  else                      v = x;
  f[i] = v;
}
__global__ void k_addcat(float* fo, long n, const float* __restrict__ cat, long row0, int tdiv) {
  long i = (long)blockIdx.x * blockDim.x + threadIdx.x;
  if (i >= n) return;
  long row = row0 + (i >> 7);
  long a = row / tdiv;
  fo[i] += cat[a * H_NUM + (i & 127)];
}
__global__ void k_add_type(float* cat, const float* __restrict__ te, const int* __restrict__ ty) {
  long i = (long)blockIdx.x * blockDim.x + threadIdx.x;
  if (i >= (long)A_NUM * H_NUM) return;
  int a = (int)(i >> 7);
  cat[i] += te[(long)ty[a] * H_NUM + (i & 127)];
}
__global__ void k_feat_agent(const float* __restrict__ pos, const float* __restrict__ head,
                             float* __restrict__ feat) {
  long i = (long)blockIdx.x * blockDim.x + threadIdx.x;
  if (i >= N_NUM) return;
  int t = (int)(i % T_NUM);
  float px = pos[i * 2], py = pos[i * 2 + 1];
  float mvx = 0.f, mvy = 0.f;
  if (t > 0) { mvx = px - pos[(i - 1) * 2]; mvy = py - pos[(i - 1) * 2 + 1]; }
  float h = head[i];
  float hx = __cosf(h), hy = __sinf(h);
  feat[i * 2 + 0] = sqrtf(mvx * mvx + mvy * mvy);
  feat[i * 2 + 1] = atan2f(hx * mvy - hy * mvx, hx * mvx + hy * mvy);
}
__global__ void k_feat_rt(const float* __restrict__ pos, const float* __restrict__ head,
                          const int* __restrict__ et, float* __restrict__ fe) {
  long i = (long)blockIdx.x * blockDim.x + threadIdx.x;
  if (i >= ET_NUM) return;
  int s = et[i], d = et[ET_NUM + i];
  float rx = pos[(long)s * 2] - pos[(long)d * 2];
  float ry = pos[(long)s * 2 + 1] - pos[(long)d * 2 + 1];
  float hd = head[d];
  float hx = __cosf(hd), hy = __sinf(hd);
  fe[i * 4 + 0] = sqrtf(rx * rx + ry * ry);
  fe[i * 4 + 1] = atan2f(hx * ry - hy * rx, hx * rx + hy * ry);
  fe[i * 4 + 2] = wrapf(head[s] - hd);
  fe[i * 4 + 3] = (float)(s - d);
}
__global__ void k_feat_a2a(const float* __restrict__ pos, const float* __restrict__ head,
                           const int* __restrict__ ea, float* __restrict__ fe) {
  long i = (long)blockIdx.x * blockDim.x + threadIdx.x;
  if (i >= EA_NUM) return;
  int si = ea[i], di = ea[EA_NUM + i];
  int as = si % A_NUM, ts = si / A_NUM;       // time-major index -> (a,t)
  int ad = di % A_NUM, td = di / A_NUM;
  long ns = (long)as * T_NUM + ts, nd = (long)ad * T_NUM + td;
  float rx = pos[ns * 2] - pos[nd * 2];
  float ry = pos[ns * 2 + 1] - pos[nd * 2 + 1];
  float hd = head[nd];
  float hx = __cosf(hd), hy = __sinf(hd);
  fe[i * 3 + 0] = sqrtf(rx * rx + ry * ry);
  fe[i * 3 + 1] = atan2f(hx * ry - hy * rx, hx * rx + hy * ry);
  fe[i * 3 + 2] = wrapf(head[ns] - hd);
}
__global__ void k_feat_pt2a(const float* __restrict__ pos, const float* __restrict__ head,
                            const float* __restrict__ ppl, const float* __restrict__ opl,
                            const int* __restrict__ ep, float* __restrict__ fe) {
  long i = (long)blockIdx.x * blockDim.x + threadIdx.x;
  if (i >= EP_NUM) return;
  int s = ep[i], di = ep[EP_NUM + i];
  int ad = di % A_NUM, td = di / A_NUM;
  long nd = (long)ad * T_NUM + td;
  float rx = ppl[(long)s * 2] - pos[nd * 2];
  float ry = ppl[(long)s * 2 + 1] - pos[nd * 2 + 1];
  float hd = head[nd];
  float hx = __cosf(hd), hy = __sinf(hd);
  fe[i * 3 + 0] = sqrtf(rx * rx + ry * ry);
  fe[i * 3 + 1] = atan2f(hx * ry - hy * rx, hx * rx + hy * ry);
  fe[i * 3 + 2] = wrapf(opl[s] - hd);
}
__global__ void k_fusion_in(const float* __restrict__ ev, const float* __restrict__ epd,
                            const float* __restrict__ ec, const int* __restrict__ ty,
                            const int* __restrict__ tokidx, const float* __restrict__ xa,
                            float* __restrict__ fin) {
  long i = (long)blockIdx.x * blockDim.x + threadIdx.x;
  if (i >= (long)N_NUM * 256) return;
  long n = i >> 8;
  int c = (int)(i & 255);
  float v;
  if (c < 128) {
    int a = (int)(n / T_NUM);
    int t2 = ty[a];
    const float* e = (t2 == 0) ? ev : (t2 == 1) ? epd : ec;
    v = e[(long)tokidx[n] * H_NUM + c];
  } else {
    v = xa[n * H_NUM + (c - 128)];
  }
  fin[i] = v;
}
// dir=0: agent-major -> time-major ; dir=1: time-major -> agent-major
__global__ void k_perm(const float* __restrict__ in, float* __restrict__ out, int dir) {
  long i = (long)blockIdx.x * blockDim.x + threadIdx.x;
  if (i >= (long)N_NUM * H_NUM) return;
  int c = (int)(i & 127);
  long n = i >> 7;
  long m;
  if (dir == 0) { long a = n / T_NUM, t = n % T_NUM; m = t * A_NUM + a; }
  else          { long t = n / A_NUM, a = n % A_NUM; m = a * T_NUM + t; }
  out[(m << 7) + c] = in[i];
}
__global__ void k_sim(const float* __restrict__ q, const float* __restrict__ k,
                      const float* __restrict__ kr, const int* __restrict__ src,
                      const int* __restrict__ dst, long E, float* __restrict__ sim) {
  long i = (long)blockIdx.x * blockDim.x + threadIdx.x;
  if (i >= E * NH_NUM) return;
  long e = i >> 3; int h = (int)(i & 7);
  const float* qp  = q  + (long)dst[e] * H_NUM + h * HD_NUM;
  const float* kp  = k  + (long)src[e] * H_NUM + h * HD_NUM;
  const float* krp = kr + e * H_NUM + h * HD_NUM;
  float s = 0.f;
#pragma unroll
  for (int j = 0; j < HD_NUM; ++j) s += qp[j] * (kp[j] + krp[j]);
  sim[i] = s * 0.25f;   // HD^-0.5
}
__global__ void k_segmax(const float* __restrict__ sim, const int* __restrict__ dst,
                         long E, unsigned* __restrict__ mkey) {
  long i = (long)blockIdx.x * blockDim.x + threadIdx.x;
  if (i >= E * NH_NUM) return;
  long e = i >> 3; int h = (int)(i & 7);
  atomicMax(&mkey[(long)dst[e] * NH_NUM + h], fenc(sim[i]));
}
__global__ void k_expden(const float* __restrict__ sim, const int* __restrict__ dst,
                         long E, const unsigned* __restrict__ mkey,
                         float* __restrict__ ex, float* __restrict__ den) {
  long i = (long)blockIdx.x * blockDim.x + threadIdx.x;
  if (i >= E * NH_NUM) return;
  long e = i >> 3; int h = (int)(i & 7);
  float m = fdec(mkey[(long)dst[e] * NH_NUM + h]);
  float x = __expf(sim[i] - m);
  ex[i] = x;
  atomicAdd(&den[(long)dst[e] * NH_NUM + h], x);
}
__global__ void k_msg(const float* __restrict__ ex, const float* __restrict__ den,
                      const float* __restrict__ v, const float* __restrict__ vr,
                      const int* __restrict__ src, const int* __restrict__ dst,
                      long E, float* __restrict__ msg) {
  long i = (long)blockIdx.x * blockDim.x + threadIdx.x;   // one thread per (e, channel)
  if (i >= E * H_NUM) return;
  long e = i >> 7; int c = (int)(i & 127); int h = c >> 4;
  long dn = dst[e], sn = src[e];
  float aw = ex[e * NH_NUM + h] / (den[dn * NH_NUM + h] + 1e-9f);
  atomicAdd(&msg[dn * H_NUM + c], aw * (v[sn * H_NUM + c] + vr[e * H_NUM + c]));
}
__global__ void k_gate(const float* __restrict__ xdst, const float* __restrict__ msg,
                       const float* __restrict__ s, const float* __restrict__ glin,
                       float* __restrict__ xo) {
  long i = (long)blockIdx.x * blockDim.x + threadIdx.x;
  if (i >= (long)N_NUM * H_NUM) return;
  float g = 1.f / (1.f + __expf(-glin[i]));
  float mm = msg[i];
  xo[i] = xdst[i] + mm + g * (s[i] - mm);
}

// ---------------------------------------------------------------------------
// host side
// ---------------------------------------------------------------------------
static inline unsigned gsz(long n) { return (unsigned)((n + 255) / 256); }

static inline void gemm(hipStream_t st, const float* Am, long lda, int M,
                        const float* Bm, long ldb, int K, int Ncol,
                        const float* bias, const float* lng, const float* lnb,
                        float* Cm, long ldc, int flags) {
  dim3 g((M + 15) / 16, (Ncol + 127) / 128);
  if (K == 128)
    k_gemm_t<128><<<g, 256, 0, st>>>(Am, lda, M, Bm, ldb, K, Ncol, bias, lng, lnb, Cm, ldc, flags);
  else
    k_gemm_t<0><<<g, 256, 0, st>>>(Am, lda, M, Bm, ldb, K, Ncol, bias, lng, lnb, Cm, ldc, flags);
}

// Input indexing: recursive insertion-order flattening of setup_inputs() dict.
// params leaves 0..317, then the 11 data arrays.
enum {
  IP_TYPE_EMB = 0, IP_SHAPE = 1, IP_XA = 7, IP_RT = 18, IP_RPT2A = 29, IP_RA2A = 40,
  IP_EMBV = 51, IP_EMBP = 57, IP_EMBC = 63, IP_FUSION = 69, IP_HEAD = 75,
  IP_TOKV = 81, IP_TOKP = 82, IP_TOKC = 83, IP_T = 84, IP_PT2A = 162, IP_A2A = 240,
  IN_POSA = 318, IN_HEADA = 319, IN_SHAPE = 320, IN_XPL = 321, IN_POSPL = 322,
  IN_ORIENT = 323, IN_ATYPE = 324, IN_ATOK = 325, IN_ET = 326, IN_EA2A = 327, IN_EPT2A = 328
};
// fourier leaf offsets: 0 freqs,1 W1,2 b1,3 ln_g,4 ln_b,5 W2,6 b2,7 out_ln_g,8 out_ln_b,9 out_W,10 out_b
// mlp: 0 W1,1 b1,2 ln_g,3 ln_b,4 W2,5 b2
// attn: 0 ln_src_g,1 ln_src_b,2 ln_dst_g,3 ln_dst_b,4 ln_r_g,5 ln_r_b,6 Wq,7 bq,8 Wk,9 bk,
//       10 Wv,11 bv,12 Wkr,13 bkr,14 Wvr,15 bvr,16 Ws,17 bs,18 Wg,19 bg,
//       20 ln_ff_g,21 ln_ff_b,22 W1,23 b1,24 W2,25 b2

struct Scratch {
  float *r_t, *r_a2a, *r_pt2a;
  float *xA, *xB, *xC, *xT1;
  float *q, *kb, *vb;
  float *sim, *ex, *mkey, *den, *msg, *sbuf, *glin, *ffh;
  float *feat, *featE, *cat, *x_a, *embv, *embp, *embc, *fin;
  float *fbuf, *hbuf, *fo, *tmp1;
  float *kr, *vr;   // overlay fbuf/hbuf (fourier finished before attention)
};

static void mlp2(hipStream_t st, const float* X, int M, int K, const float* const* P, int pb,
                 int dout, float* hid, float* out) {
  gemm(st, X, K, M, P[pb + 0], H_NUM, K, H_NUM, P[pb + 1], 0, 0, hid, H_NUM, 0);
  gemm(st, hid, H_NUM, M, P[pb + 4], dout, H_NUM, dout, P[pb + 5],
       P[pb + 2], P[pb + 3], out, dout, GF_LN | GF_ARELU);
}

static void fourier(hipStream_t st, const float* X, long rows, int din,
                    const float* const* P, int pb, const float* catv, float* R, Scratch& w) {
  const float* freqs = P[pb + 0];
  const float* W1 = P[pb + 1];  const float* b1 = P[pb + 2];
  const float* lg = P[pb + 3];  const float* lb = P[pb + 4];
  const float* W2 = P[pb + 5];  const float* b2 = P[pb + 6];
  const float* og = P[pb + 7];  const float* ob2 = P[pb + 8];
  const float* oW = P[pb + 9];  const float* obb = P[pb + 10];
  for (long c0 = 0; c0 < rows; c0 += CH_NUM) {
    long cr = rows - c0; if (cr > CH_NUM) cr = CH_NUM;
    long tot = cr * din * 129;
    k_fgen<<<gsz(tot), 256, 0, st>>>(X + c0 * din, cr, din, freqs, w.fbuf);
    for (int d = 0; d < din; ++d)                   // h_d = f_d @ W1_d + b1_d
      gemm(st, w.fbuf + d * 129, (long)din * 129, (int)cr,
           W1 + (long)d * 129 * H_NUM, H_NUM, 129, H_NUM,
           b1 + d * H_NUM, 0, 0, w.hbuf + d * H_NUM, (long)din * H_NUM, 0);
    for (int d = 0; d < din; ++d)                   // fo = sum_d relu(LN(h_d)) @ W2_d + b2_d
      gemm(st, w.hbuf + d * H_NUM, (long)din * H_NUM, (int)cr,
           W2 + (long)d * H_NUM * H_NUM, H_NUM, H_NUM, H_NUM,
           b2 + d * H_NUM, lg + d * H_NUM, lb + d * H_NUM,
           w.fo, H_NUM, GF_LN | GF_ARELU | (d ? GF_ACC : 0));
    if (catv)
      k_addcat<<<gsz(cr * H_NUM), 256, 0, st>>>(w.fo, cr * H_NUM, catv, c0, T_NUM);
    gemm(st, w.fo, H_NUM, (int)cr, oW, H_NUM, H_NUM, H_NUM,
         obb, og, ob2, R + c0 * H_NUM, H_NUM, GF_LN | GF_ARELU);
  }
}

static void run_attn(hipStream_t st, const float* const* P, int pb,
                     const float* xsrc, int Msrc, const float* xdst,
                     const float* r, const int* esrc, const int* edst, long E,
                     float* out, Scratch& w) {
  // projections (LN fused into A staging)
  gemm(st, xdst, H_NUM, N_NUM, P[pb + 6],  H_NUM, H_NUM, H_NUM, P[pb + 7],
       P[pb + 2], P[pb + 3], w.q,  H_NUM, GF_LN);
  gemm(st, xsrc, H_NUM, Msrc,  P[pb + 8],  H_NUM, H_NUM, H_NUM, P[pb + 9],
       P[pb + 0], P[pb + 1], w.kb, H_NUM, GF_LN);
  gemm(st, xsrc, H_NUM, Msrc,  P[pb + 10], H_NUM, H_NUM, H_NUM, P[pb + 11],
       P[pb + 0], P[pb + 1], w.vb, H_NUM, GF_LN);
  gemm(st, r, H_NUM, (int)E,   P[pb + 12], H_NUM, H_NUM, H_NUM, P[pb + 13],
       P[pb + 4], P[pb + 5], w.kr, H_NUM, GF_LN);
  gemm(st, r, H_NUM, (int)E,   P[pb + 14], H_NUM, H_NUM, H_NUM, P[pb + 15],
       P[pb + 4], P[pb + 5], w.vr, H_NUM, GF_LN);
  // segment softmax + weighted aggregation
  long EH = E * NH_NUM;
  k_fill<<<gsz((long)N_NUM * NH_NUM), 256, 0, st>>>(w.mkey, (long)N_NUM * NH_NUM, 0.f);
  k_fill<<<gsz((long)N_NUM * NH_NUM), 256, 0, st>>>(w.den,  (long)N_NUM * NH_NUM, 0.f);
  k_fill<<<gsz((long)N_NUM * H_NUM), 256, 0, st>>>(w.msg,  (long)N_NUM * H_NUM, 0.f);
  k_sim<<<gsz(EH), 256, 0, st>>>(w.q, w.kb, w.kr, esrc, edst, E, w.sim);
  k_segmax<<<gsz(EH), 256, 0, st>>>(w.sim, edst, E, (unsigned*)w.mkey);
  k_expden<<<gsz(EH), 256, 0, st>>>(w.sim, edst, E, (const unsigned*)w.mkey, w.ex, w.den);
  k_msg<<<gsz(E * H_NUM), 256, 0, st>>>(w.ex, w.den, w.vb, w.vr, esrc, edst, E, w.msg);
  // gating: g = sigmoid([msg, LN(xd)] @ Wg + bg); x1 = xd + msg + g*(LN(xd)@Ws+bs - msg)
  gemm(st, xdst, H_NUM, N_NUM, P[pb + 16], H_NUM, H_NUM, H_NUM, P[pb + 17],
       P[pb + 2], P[pb + 3], w.sbuf, H_NUM, GF_LN);
  gemm(st, w.msg, H_NUM, N_NUM, P[pb + 18], H_NUM, H_NUM, H_NUM, P[pb + 19],
       0, 0, w.glin, H_NUM, 0);
  gemm(st, xdst, H_NUM, N_NUM, P[pb + 18] + (long)H_NUM * H_NUM, H_NUM, H_NUM, H_NUM,
       0, P[pb + 2], P[pb + 3], w.glin, H_NUM, GF_LN | GF_ACC);
  k_gate<<<gsz((long)N_NUM * H_NUM), 256, 0, st>>>(xdst, w.msg, w.sbuf, w.glin, w.xT1);
  // FFN: out = x1 + relu(LN(x1)@W1+b1) @ W2 + b2
  gemm(st, w.xT1, H_NUM, N_NUM, P[pb + 22], FF_NUM, H_NUM, FF_NUM, P[pb + 23],
       P[pb + 20], P[pb + 21], w.ffh, FF_NUM, GF_LN | GF_CRELU);
  k_copy<<<gsz((long)N_NUM * H_NUM), 256, 0, st>>>(out, w.xT1, (long)N_NUM * H_NUM);
  gemm(st, w.ffh, FF_NUM, N_NUM, P[pb + 24], H_NUM, FF_NUM, H_NUM, P[pb + 25],
       0, 0, out, H_NUM, GF_ACC);
}

extern "C" void kernel_launch(void* const* d_in, const int* in_sizes, int n_in,
                              void* d_out, int out_size, void* d_ws, size_t ws_size,
                              hipStream_t stream) {
  const float* P[330];
  for (int i = 0; i < 329 && i < n_in; ++i) P[i] = (const float*)d_in[i];
  const float* pos_a  = P[IN_POSA];
  const float* head_a = P[IN_HEADA];
  const float* shp    = P[IN_SHAPE];
  const float* x_pl   = P[IN_XPL];
  const float* pos_pl = P[IN_POSPL];
  const float* orient = P[IN_ORIENT];
  const int* atype = (const int*)d_in[IN_ATYPE];
  const int* atok  = (const int*)d_in[IN_ATOK];
  const int* et    = (const int*)d_in[IN_ET];
  const int* ea    = (const int*)d_in[IN_EA2A];
  const int* ep    = (const int*)d_in[IN_EPT2A];

  // ---- workspace layout (floats) ----
  Scratch w;
  float* p = (float*)d_ws;
  const long NB = (long)N_NUM * H_NUM;
  w.r_t    = p; p += (long)ET_NUM * H_NUM;
  w.r_a2a  = p; p += (long)EA_NUM * H_NUM;
  w.r_pt2a = p; p += (long)EP_NUM * H_NUM;
  w.xA = p; p += NB;  w.xB = p; p += NB;  w.xC = p; p += NB;  w.xT1 = p; p += NB;
  w.q  = p; p += NB;
  w.kb = p; p += (long)MS_NUM * H_NUM;
  w.vb = p; p += (long)MS_NUM * H_NUM;
  w.sim = p; p += (long)EA_NUM * NH_NUM;
  w.ex  = p; p += (long)EA_NUM * NH_NUM;
  w.mkey = p; p += (long)N_NUM * NH_NUM;
  w.den  = p; p += (long)N_NUM * NH_NUM;
  w.msg  = p; p += NB;  w.sbuf = p; p += NB;  w.glin = p; p += NB;
  w.ffh  = p; p += (long)N_NUM * FF_NUM;
  w.feat = p; p += (long)N_NUM * 2;
  w.featE = p; p += (long)EA_NUM * 4;
  w.cat  = p; p += (long)A_NUM * H_NUM;
  w.x_a  = p; p += NB;
  w.embv = p; p += (long)TOK_NUM * H_NUM;
  w.embp = p; p += (long)TOK_NUM * H_NUM;
  w.embc = p; p += (long)TOK_NUM * H_NUM;
  w.fin  = p; p += (long)N_NUM * 256;
  w.fbuf = p; p += (long)CH_NUM * 4 * 129;   // fourier f chunk; later overlaid by kr
  w.hbuf = p; p += (long)CH_NUM * 4 * H_NUM; // fourier h chunk; later overlaid by vr
  w.fo   = p; p += (long)CH_NUM * H_NUM;
  w.tmp1 = p; p += (long)A_NUM * H_NUM;
  w.kr = w.fbuf;  // E*128 <= CH*4*129
  w.vr = w.hbuf;  // E*128 <= CH*4*128

  // ---- agent features, categorical embedding ----
  k_feat_agent<<<gsz(N_NUM), 256, 0, stream>>>(pos_a, head_a, w.feat);
  mlp2(stream, shp, A_NUM, 3, P, IP_SHAPE, H_NUM, w.tmp1, w.cat);
  k_add_type<<<gsz((long)A_NUM * H_NUM), 256, 0, stream>>>(w.cat, P[IP_TYPE_EMB], atype);
  fourier(stream, w.feat, N_NUM, 2, P, IP_XA, w.cat, w.x_a, w);

  // ---- token embeddings + fusion ----
  mlp2(stream, P[IP_TOKV], TOK_NUM, 8, P, IP_EMBV, H_NUM, w.tmp1, w.embv);
  mlp2(stream, P[IP_TOKP], TOK_NUM, 8, P, IP_EMBP, H_NUM, w.tmp1, w.embp);
  mlp2(stream, P[IP_TOKC], TOK_NUM, 8, P, IP_EMBC, H_NUM, w.tmp1, w.embc);
  k_fusion_in<<<gsz((long)N_NUM * 256), 256, 0, stream>>>(w.embv, w.embp, w.embc,
                                                          atype, atok, w.x_a, w.fin);
  mlp2(stream, w.fin, N_NUM, 256, P, IP_FUSION, H_NUM, w.ffh, w.xA);

  // ---- relative edge embeddings (fourier) ----
  k_feat_rt<<<gsz(ET_NUM), 256, 0, stream>>>(pos_a, head_a, et, w.featE);
  fourier(stream, w.featE, ET_NUM, 4, P, IP_RT, 0, w.r_t, w);
  k_feat_a2a<<<gsz(EA_NUM), 256, 0, stream>>>(pos_a, head_a, ea, w.featE);
  fourier(stream, w.featE, EA_NUM, 3, P, IP_RA2A, 0, w.r_a2a, w);
  k_feat_pt2a<<<gsz(EP_NUM), 256, 0, stream>>>(pos_a, head_a, pos_pl, orient, ep, w.featE);
  fourier(stream, w.featE, EP_NUM, 3, P, IP_RPT2A, 0, w.r_pt2a, w);

  // ---- attention layers ----
  for (int i = 0; i < L_NUM; ++i) {
    run_attn(stream, P, IP_T + i * 26,    w.xA, N_NUM, w.xA, w.r_t,
             et, et + ET_NUM, ET_NUM, w.xB, w);                       // temporal (agent-major)
    k_perm<<<gsz(NB), 256, 0, stream>>>(w.xB, w.xC, 0);               // -> time-major
    run_attn(stream, P, IP_PT2A + i * 26, x_pl, MS_NUM, w.xC, w.r_pt2a,
             ep, ep + EP_NUM, EP_NUM, w.xB, w);                       // map -> agent
    run_attn(stream, P, IP_A2A + i * 26,  w.xB, N_NUM, w.xB, w.r_a2a,
             ea, ea + EA_NUM, EA_NUM, w.xC, w);                       // agent -> agent
    k_perm<<<gsz(NB), 256, 0, stream>>>(w.xC, w.xA, 1);               // -> agent-major
  }

  // ---- head MLP -> d_out (A,T,TOK) f32 ----
  mlp2(stream, w.xA, N_NUM, H_NUM, P, IP_HEAD, TOK_NUM, w.q, (float*)d_out);
}
